// GCN_46566035423458
// MI455X (gfx1250) — matrix operand order repaired and checked by reading, not compiled, not run on hardware
//
#include <hip/hip_runtime.h>
#include <hip/hip_bf16.h>
#include <math.h>

typedef _Float16 f16;
typedef __attribute__((ext_vector_type(16))) _Float16 v16h;
typedef __attribute__((ext_vector_type(8)))  _Float16 v8h;
typedef __attribute__((ext_vector_type(8)))  float    v8f;

#define BATCH 2
#define C_IN  256
#define C_MID 128
#define HW    9216      // 96*96
#define C_CAT 512
#define BN_CNT ((float)(BATCH * HW))

__device__ __forceinline__ v16h hcat(v8h a, v8h b) {
  return __builtin_shufflevector(a, b, 0,1,2,3,4,5,6,7,8,9,10,11,12,13,14,15);
}

// A-operand (16x32 f16) gather from row-major [row][K] storage.
// ISA 16-bit A layout: lane L, row = L%16; halves K = base..base+7 and
// 16+base..16+base+7 with base = (L>=16)*8  -> two 16B loads.
__device__ __forceinline__ v16h load_A(const f16* __restrict__ base, int row0, int ld, int k0) {
  const int lane = threadIdx.x & 31;
  const int r    = lane & 15;
  const int sel  = (lane >> 4) & 1;
  const f16* p = base + (size_t)(row0 + r) * ld + k0 + sel * 8;
  v8h lo = *(const v8h*)p;
  v8h hi = *(const v8h*)(p + 16);
  return hcat(lo, hi);
}

// B-operand (32x16 f16) gather from K-contiguous-per-column [col][K] storage.
// ISA 16-bit B layout: lane L, col = L%16; K = sel*8 + 0..7 and 16+sel*8 + 0..7 -> two 16B loads.
__device__ __forceinline__ v16h load_B(const f16* __restrict__ base, int col0, size_t ld, int k0) {
  const int lane = threadIdx.x & 31;
  const int c    = lane & 15;
  const int sel  = (lane >> 4) & 1;
  const f16* p = base + (size_t)(col0 + c) * ld + k0 + sel * 8;
  v8h lo = *(const v8h*)p;
  v8h hi = *(const v8h*)(p + 16);
  return hcat(lo, hi);
}

__device__ __forceinline__ v8f wmma16(v16h a, v16h b, v8f c) {
  return __builtin_amdgcn_wmma_f32_16x16x32_f16(false, a, false, b, (short)0, c, false, false);
}

// ---------------- pre-pass: transpose/convert ----------------

// x [b][256][N] f32 -> xT [b][N][256] f16
__global__ void k_transpose_x(const float* __restrict__ x, f16* __restrict__ xT) {
  size_t idx = (size_t)blockIdx.x * blockDim.x + threadIdx.x;
  const size_t total = (size_t)BATCH * C_IN * HW;
  if (idx >= total) return;
  int n = (int)(idx % HW);
  int c = (int)((idx / HW) % C_IN);
  int b = (int)(idx / ((size_t)HW * C_IN));
  xT[((size_t)b * HW + n) * C_IN + c] = (f16)x[idx];
}

__global__ void k_convert_w(const float* __restrict__ src, f16* __restrict__ dst, int n, float scale) {
  int i = blockIdx.x * blockDim.x + threadIdx.x;
  if (i < n) dst[i] = (f16)(src[i] * scale);
}

// ---------------- projection GEMMs (q,k,v) ----------------
// D[16 o x 16 n] = W[128x256] * x[256xN] tile; K = 256 -> 8 WMMAs.
__global__ void __launch_bounds__(128) k_proj(const f16* __restrict__ Wh, const f16* __restrict__ xT,
                                              f16* __restrict__ outT, f16* __restrict__ outN,
                                              int transposed) {
  const int lane = threadIdx.x & 31;
  const int wid  = threadIdx.x >> 5;
  const int b    = blockIdx.z;
  const int o0   = blockIdx.y * 16;
  const int n0   = (blockIdx.x * 4 + wid) * 16;
  const f16* xb = xT + (size_t)b * HW * C_IN;
  v8f acc = {0.f,0.f,0.f,0.f,0.f,0.f,0.f,0.f};
#pragma unroll
  for (int j = 0; j < C_IN; j += 32)
    acc = wmma16(load_A(Wh, o0, C_IN, j), load_B(xb, n0, C_IN, j), acc);
  const int col = lane & 15;
  const int sel = (lane >> 4) & 1;
  if (transposed) {
    // outT [b][N][128]: lane owns 8 consecutive channels of row n0+col -> one b128 store
    v8h h = {};
#pragma unroll
    for (int v = 0; v < 8; ++v) h[v] = (f16)acc[v];
    f16* p = outT + ((size_t)b * HW + n0 + col) * C_MID + o0 + sel * 8;
    *(v8h*)p = h;
  } else {
    // outN [b][128][N] natural layout (B-operand layout for attn @ v)
    f16* p = outN + (size_t)b * C_MID * HW;
#pragma unroll
    for (int v = 0; v < 8; ++v)
      p[(size_t)(o0 + sel * 8 + v) * HW + n0 + col] = (f16)acc[v];
  }
}

// ---------------- flash attention ----------------
// One wave per 16-row query block; loop all 9216 keys, 32 at a time.
__global__ void __launch_bounds__(128) k_flash(const f16* __restrict__ qT, const f16* __restrict__ kT,
                                               const f16* __restrict__ vcm, f16* __restrict__ gcnT) {
  __shared__ f16 pbuf[4][16 * 32];           // per-wave P-tile transpose buffer
  const int lane = threadIdx.x & 31;
  const int wid  = threadIdx.x >> 5;
  const int b    = blockIdx.y;
  const int n0   = (blockIdx.x * 4 + wid) * 16;
  const f16* qb = qT  + (size_t)b * HW * C_MID;
  const f16* kb = kT  + (size_t)b * HW * C_MID;
  const f16* vb = vcm + (size_t)b * C_MID * HW;

  v16h aQ[4];
#pragma unroll
  for (int j = 0; j < 4; ++j) aQ[j] = load_A(qb, n0, C_MID, 32 * j);

  v8f o_acc[8];
#pragma unroll
  for (int t = 0; t < 8; ++t) o_acc[t] = (v8f){0.f,0.f,0.f,0.f,0.f,0.f,0.f,0.f};
  float m_i[8], l_i[8];
#pragma unroll
  for (int v = 0; v < 8; ++v) { m_i[v] = -3.0e38f; l_i[v] = 0.f; }

  f16* pw = pbuf[wid];
  const int col = lane & 15;
  const int sel = (lane >> 4) & 1;

  for (int m0 = 0; m0 < HW; m0 += 32) {
    if (m0 + 32 < HW) {
      __builtin_prefetch(kb + (size_t)(m0 + 32 + col) * C_MID, 0, 1);
    }
    // S = Q K^T tiles (16x16 x2), K-dim = 128
    v8f s0 = {0.f,0.f,0.f,0.f,0.f,0.f,0.f,0.f};
    v8f s1 = {0.f,0.f,0.f,0.f,0.f,0.f,0.f,0.f};
#pragma unroll
    for (int j = 0; j < 4; ++j) {
      v16h b0 = load_B(kb, m0,      C_MID, 32 * j);
      v16h b1 = load_B(kb, m0 + 16, C_MID, 32 * j);
      s0 = wmma16(aQ[j], b0, s0);
      s1 = wmma16(aQ[j], b1, s1);
    }
    // online softmax: row r = v + sel*8 lives in 16 lanes of this half-wave
    float mx[8], p0[8], p1[8], rs[8], alpha[8];
#pragma unroll
    for (int v = 0; v < 8; ++v) mx[v] = fmaxf(s0[v], s1[v]);
    for (int off = 1; off < 16; off <<= 1)
#pragma unroll
      for (int v = 0; v < 8; ++v) mx[v] = fmaxf(mx[v], __shfl_xor(mx[v], off, 32));
#pragma unroll
    for (int v = 0; v < 8; ++v) {
      float mn = fmaxf(m_i[v], mx[v]);
      alpha[v] = __expf(m_i[v] - mn);
      p0[v] = __expf(s0[v] - mn);
      p1[v] = __expf(s1[v] - mn);
      rs[v] = p0[v] + p1[v];
      m_i[v] = mn;
    }
    for (int off = 1; off < 16; off <<= 1)
#pragma unroll
      for (int v = 0; v < 8; ++v) rs[v] += __shfl_xor(rs[v], off, 32);
#pragma unroll
    for (int v = 0; v < 8; ++v) l_i[v] = l_i[v] * alpha[v] + rs[v];
#pragma unroll
    for (int t = 0; t < 8; ++t)
#pragma unroll
      for (int v = 0; v < 8; ++v) o_acc[t][v] *= alpha[v];

    // transpose P (lane-owns-column -> lane-owns-row) via LDS; same-wave DS is in-order
#pragma unroll
    for (int v = 0; v < 8; ++v) {
      int row = v + sel * 8;
      pw[row * 32 + col]      = (f16)p0[v];
      pw[row * 32 + 16 + col] = (f16)p1[v];
    }
    v16h aP = load_A(pw, 0, 32, 0);
    // O += P V^T, c-dim in 8 tiles of 16
#pragma unroll
    for (int t = 0; t < 8; ++t) {
      v16h bV = load_B(vb, 16 * t, (size_t)HW, m0);
      o_acc[t] = wmma16(aP, bV, o_acc[t]);
    }
  }
  // normalize rows by l and store gcnT [b][N][128] f16
  float inv_l[8];
#pragma unroll
  for (int v = 0; v < 8; ++v) inv_l[v] = 1.f / l_i[v];
#pragma unroll
  for (int t = 0; t < 8; ++t)
#pragma unroll
    for (int v = 0; v < 8; ++v)
      gcnT[((size_t)b * HW + n0 + v + sel * 8) * C_MID + 16 * t + col] =
          (f16)(o_acc[t][v] * inv_l[v]);
}

// ---------------- conv GEMM + BN batch-stat accumulation ----------------
__global__ void __launch_bounds__(128) k_gemm_stats(const f16* __restrict__ Wh, const f16* __restrict__ BT,
                                                    float* __restrict__ outp,
                                                    float* __restrict__ sum, float* __restrict__ sq,
                                                    int K) {
  const int lane = threadIdx.x & 31;
  const int wid  = threadIdx.x >> 5;
  const int b    = blockIdx.z;
  const int o0   = blockIdx.y * 16;
  const int n0   = (blockIdx.x * 4 + wid) * 16;
  const f16* bb = BT + (size_t)b * HW * K;
  v8f acc = {0.f,0.f,0.f,0.f,0.f,0.f,0.f,0.f};
  for (int j = 0; j < K; j += 32)
    acc = wmma16(load_A(Wh, o0, K, j), load_B(bb, n0, (size_t)K, j), acc);
  const int col = lane & 15;
  const int sel = (lane >> 4) & 1;
  float* po = outp + (size_t)b * C_IN * HW;
  float rs[8], rq[8];
#pragma unroll
  for (int v = 0; v < 8; ++v) {
    float val = acc[v];
    po[(size_t)(o0 + sel * 8 + v) * HW + n0 + col] = val;
    rs[v] = val;
    rq[v] = val * val;
  }
  for (int off = 1; off < 16; off <<= 1)
#pragma unroll
    for (int v = 0; v < 8; ++v) {
      rs[v] += __shfl_xor(rs[v], off, 32);
      rq[v] += __shfl_xor(rq[v], off, 32);
    }
  if (col == 0) {  // lanes 0 and 16 cover the two row-halves
#pragma unroll
    for (int v = 0; v < 8; ++v) {
      atomicAdd(&sum[o0 + sel * 8 + v], rs[v]);
      atomicAdd(&sq[o0 + sel * 8 + v],  rq[v]);
    }
  }
}

// ---------------- BN1 + ReLU + concat -> catT [b][N][512] f16 ----------------
__global__ void k_bn_cat(const float* __restrict__ g1, const f16* __restrict__ xT,
                         const float* __restrict__ sum, const float* __restrict__ sq,
                         const float* __restrict__ gamma, const float* __restrict__ beta,
                         f16* __restrict__ catT) {
  size_t idx = (size_t)blockIdx.x * blockDim.x + threadIdx.x;
  const size_t total = (size_t)BATCH * HW * C_CAT;
  if (idx >= total) return;
  int c2 = (int)(idx % C_CAT);
  size_t bn = idx / C_CAT;               // b*HW + n
  f16 outv;
  if (c2 < C_IN) {
    outv = xT[bn * C_IN + c2];
  } else {
    int o = c2 - C_IN;
    int b = (int)(bn / HW);
    int n = (int)(bn % HW);
    float mean = sum[o] / BN_CNT;
    float var  = sq[o] / BN_CNT - mean * mean;
    float rstd = rsqrtf(var + 1e-5f);
    float v = g1[((size_t)b * C_IN + o) * HW + n];
    float y = (v - mean) * rstd * gamma[o] + beta[o];
    outv = (f16)fmaxf(y, 0.f);
  }
  catT[idx] = outv;
}

// ---------------- final BN2 + ReLU -> d_out f32 ----------------
__global__ void k_bn_out(const float* __restrict__ pre,
                         const float* __restrict__ sum, const float* __restrict__ sq,
                         const float* __restrict__ gamma, const float* __restrict__ beta,
                         float* __restrict__ out) {
  size_t idx = (size_t)blockIdx.x * blockDim.x + threadIdx.x;
  const size_t total = (size_t)BATCH * C_IN * HW;
  if (idx >= total) return;
  int o = (int)((idx / HW) % C_IN);
  float mean = sum[o] / BN_CNT;
  float var  = sq[o] / BN_CNT - mean * mean;
  float rstd = rsqrtf(var + 1e-5f);
  float y = (pre[idx] - mean) * rstd * gamma[o] + beta[o];
  out[idx] = fmaxf(y, 0.f);
}

extern "C" void kernel_launch(void* const* d_in, const int* in_sizes, int n_in,
                              void* d_out, int out_size, void* d_ws, size_t ws_size,
                              hipStream_t stream) {
  const float* x      = (const float*)d_in[0];
  const float* Wk     = (const float*)d_in[1];
  const float* Wv     = (const float*)d_in[2];
  const float* Wq     = (const float*)d_in[3];
  const float* Wc1    = (const float*)d_in[4];
  const float* gamma1 = (const float*)d_in[5];
  const float* beta1  = (const float*)d_in[6];
  const float* Wout   = (const float*)d_in[7];
  const float* gamma2 = (const float*)d_in[8];
  const float* beta2  = (const float*)d_in[9];
  float* out = (float*)d_out;
  char* w = (char*)d_ws;

  // workspace layout (bytes)
  const size_t off_xT    = 0;                       // 2*9216*256*2  = 9,437,184
  const size_t off_Wqh   = off_xT  + 9437184;       // 65,536
  const size_t off_Wkh   = off_Wqh + 65536;
  const size_t off_Wvh   = off_Wkh + 65536;
  const size_t off_Wc1h  = off_Wvh + 65536;
  const size_t off_Wouth = off_Wc1h + 65536;        // 262,144
  const size_t off_BB    = off_Wouth + 262144;      // big block 18,874,368
  const size_t off_qT    = off_BB;                  // 4,718,592 each
  const size_t off_kT    = off_BB + 4718592;
  const size_t off_vcm   = off_BB + 2 * 4718592;
  const size_t off_gcnT  = off_BB + 3 * 4718592;
  const size_t off_outp  = off_BB;                  // aliased after attention retires
  const size_t off_g1    = off_BB + 18874368;       // 18,874,368
  const size_t off_catT  = off_g1 + 18874368;       // 18,874,368
  const size_t off_stats = off_catT + 18874368;     // 4 * 256 * 4 = 4096

  f16*   xT    = (f16*)(w + off_xT);
  f16*   Wqh   = (f16*)(w + off_Wqh);
  f16*   Wkh   = (f16*)(w + off_Wkh);
  f16*   Wvh   = (f16*)(w + off_Wvh);
  f16*   Wc1h  = (f16*)(w + off_Wc1h);
  f16*   Wouth = (f16*)(w + off_Wouth);
  f16*   qT    = (f16*)(w + off_qT);
  f16*   kT    = (f16*)(w + off_kT);
  f16*   vcm   = (f16*)(w + off_vcm);
  f16*   gcnT  = (f16*)(w + off_gcnT);
  float* outp  = (float*)(w + off_outp);
  float* g1    = (float*)(w + off_g1);
  f16*   catT  = (f16*)(w + off_catT);
  float* sum1  = (float*)(w + off_stats);
  float* sq1   = sum1 + 256;
  float* sum2  = sq1 + 256;
  float* sq2   = sum2 + 256;

  hipMemsetAsync(w + off_stats, 0, 4096, stream);

  // transpose/convert x; convert weights (fold 1/sqrt(128) into Wq)
  k_transpose_x<<<dim3((BATCH * C_IN * HW + 255) / 256), dim3(256), 0, stream>>>(x, xT);
  const float qscale = 0.08838834764831845f;  // 128^-0.5
  k_convert_w<<<dim3((32768 + 255) / 256), dim3(256), 0, stream>>>(Wq, Wqh, 32768, qscale);
  k_convert_w<<<dim3((32768 + 255) / 256), dim3(256), 0, stream>>>(Wk, Wkh, 32768, 1.f);
  k_convert_w<<<dim3((32768 + 255) / 256), dim3(256), 0, stream>>>(Wv, Wvh, 32768, 1.f);
  k_convert_w<<<dim3((32768 + 255) / 256), dim3(256), 0, stream>>>(Wc1, Wc1h, 32768, 1.f);
  k_convert_w<<<dim3((131072 + 255) / 256), dim3(256), 0, stream>>>(Wout, Wouth, 131072, 1.f);

  // q,k,v projections (WMMA)
  k_proj<<<dim3(144, 8, 2), dim3(128), 0, stream>>>(Wqh, xT, qT, nullptr, 1);
  k_proj<<<dim3(144, 8, 2), dim3(128), 0, stream>>>(Wkh, xT, kT, nullptr, 1);
  k_proj<<<dim3(144, 8, 2), dim3(128), 0, stream>>>(Wvh, xT, nullptr, vcm, 0);

  // flash attention (WMMA + online softmax)
  k_flash<<<dim3(144, 2), dim3(128), 0, stream>>>(qT, kT, vcm, gcnT);

  // conv1 GEMM + BN1 stats
  k_gemm_stats<<<dim3(144, 16, 2), dim3(128), 0, stream>>>(Wc1h, gcnT, g1, sum1, sq1, C_MID);
  // BN1 + ReLU + concat with x
  k_bn_cat<<<dim3((BATCH * HW * C_CAT + 255) / 256), dim3(256), 0, stream>>>(
      g1, xT, sum1, sq1, gamma1, beta1, catT);
  // out conv GEMM + BN2 stats (outp aliases retired q/k/v/gcn block)
  k_gemm_stats<<<dim3(144, 16, 2), dim3(128), 0, stream>>>(Wouth, catT, outp, sum2, sq2, C_CAT);
  // final BN2 + ReLU
  k_bn_out<<<dim3((BATCH * C_IN * HW + 255) / 256), dim3(256), 0, stream>>>(
      outp, sum2, sq2, gamma2, beta2, out);
}
